// AttentionFusion_6640019440077
// MI455X (gfx1250) — matrix operand order repaired and checked by reading, not compiled, not run on hardware
//
#include <hip/hip_runtime.h>
#include <hip/hip_bf16.h>

#define DD 768
#define HD 768
#define NHEAD 4
#define BB 4
#define T1 128
#define T2 128
#define LN_EPS 1e-5f

typedef __attribute__((ext_vector_type(16))) __bf16          v16bf;
typedef __attribute__((ext_vector_type(8)))  float           v8f;

struct U32x8 { uint4 lo; uint4 hi; };   // 32 bytes == one 16x16x32 bf16 fragment

__device__ __forceinline__ unsigned short f2bf(float f) {
    unsigned int u; __builtin_memcpy(&u, &f, 4);
    u += 0x7fffu + ((u >> 16) & 1u);          // round-to-nearest-even
    return (unsigned short)(u >> 16);
}

// ---------------------------------------------------------------- converts
__global__ void f2bf4_kernel(const float4* __restrict__ s, ushort4* __restrict__ d, int n4) {
    int i = blockIdx.x * blockDim.x + threadIdx.x;
    if (i < n4) {
        float4 v = s[i];
        ushort4 o;
        o.x = f2bf(v.x); o.y = f2bf(v.y); o.z = f2bf(v.z); o.w = f2bf(v.w);
        d[i] = o;
    }
}

// fused[:, 0:768] = speech (bf16), 4 elems/thread
__global__ void pack_speech_kernel(const uint2* __restrict__ sB, uint2* __restrict__ fused) {
    int i = blockIdx.x * blockDim.x + threadIdx.x;     // over 512*192 uint2s
    if (i < BB * T1 * (DD / 4)) {
        int r = i / (DD / 4), c = i % (DD / 4);
        fused[(size_t)r * (2 * DD / 4) + c] = sB[i];
    }
}

// --------------------------------------- tiled transpose + f32->bf16 convert
// X[R][C] f32  ->  Xt[C][R] bf16   (batched via blockIdx.z)
__global__ void transpose_f2bf(const float* __restrict__ X, unsigned short* __restrict__ Xt,
                               int R, int C, long long sX, long long sXt) {
    __shared__ unsigned short tile[32][33];
    const float* Xz = X + (size_t)blockIdx.z * sX;
    unsigned short* Xtz = Xt + (size_t)blockIdx.z * sXt;
    const int c0 = blockIdx.x * 32, r0 = blockIdx.y * 32;
    const int tx = threadIdx.x, ty = threadIdx.y;       // 32 x 8
#pragma unroll
    for (int i = 0; i < 32; i += 8)
        tile[ty + i][tx] = f2bf(Xz[(size_t)(r0 + ty + i) * C + (c0 + tx)]);
    __syncthreads();
#pragma unroll
    for (int i = 0; i < 32; i += 8)
        Xtz[(size_t)(c0 + ty + i) * R + (r0 + tx)] = tile[tx][ty + i];
}

// ---------------------------------------------------------------- WMMA GEMM
// C[M,N] = A[M,K](bf16 row-major) x Bt[N,K](bf16 pre-transposed) (+bias)(+relu)
// mode 0: f32 row-major   mode 1: bf16 row-major
// mode 2: f32 transposed w/ b-split: out[(row>>7)][col][row&127]   (Vt)
// Each wave: 32x64 tile = 2 A-frags x 4 B-frags -> 8 WMMAs / 32-deep k-step.
// Single fragment buffer (no explicit double-buffer: avoids VGPR spills);
// base pointers hoisted so the k-loop uses pure immediate-offset b128 loads.
__global__ void __launch_bounds__(256)
wmma_gemm_bf16(const unsigned short* __restrict__ A,
               const unsigned short* __restrict__ Bt,
               const float* __restrict__ bias,
               float* __restrict__ outF,
               unsigned short* __restrict__ outB,
               int M, int N, int K, int lda, int ldbt, int ldc,
               long long strideA, long long strideB,
               long long strideBias, long long strideOut,
               int mode, int doBias, int doRelu) {
    const int z = blockIdx.z;
    A  += (size_t)z * strideA;
    Bt += (size_t)z * strideB;
    if (doBias) bias += (size_t)z * strideBias;
    if (outF) outF += (size_t)z * strideOut;
    if (outB) outB += (size_t)z * strideOut;

    const int tilesN = N >> 6;                          // 64-wide strips
    const int nTiles = (M >> 5) * tilesN;               // 32-row blocks
    const int tileId = blockIdx.x * (blockDim.x >> 5) + (threadIdx.x >> 5);
    if (tileId >= nTiles) return;                       // wave-uniform exit
    const int tm = tileId / tilesN;
    const int tn = tileId % tilesN;

    const int lane   = threadIdx.x & 31;
    const int laneLo = lane & 15;
    const int laneHi = lane >> 4;

    // A frag (ISA 7.12.2): row m=laneLo; elems 0..7 -> K=laneHi*8+e,
    // elems 8..15 -> K=16+laneHi*8+e  => two contiguous 16B loads per lane.
    const unsigned short* Ar0 = A + (size_t)(tm * 32 + laneLo) * lda + laneHi * 8;
    const unsigned short* Ar1 = Ar0 + (size_t)16 * lda;
    // B frag: col n=laneLo; elems 0..7 -> K=laneHi*8+e, elems 8..15 -> K=16+laneHi*8+e.
    const unsigned short* Bp[4];
#pragma unroll
    for (int j = 0; j < 4; ++j)
        Bp[j] = Bt + (size_t)(tn * 64 + j * 16 + laneLo) * ldbt + laneHi * 8;

    v8f acc[2][4] = {};

    for (int kb = 0; kb < K; kb += 32) {
        U32x8 ta0, ta1;
        ta0.lo = *(const uint4*)(Ar0 + kb);
        ta0.hi = *(const uint4*)(Ar0 + kb + 16);
        ta1.lo = *(const uint4*)(Ar1 + kb);
        ta1.hi = *(const uint4*)(Ar1 + kb + 16);
        const v16bf af0 = __builtin_bit_cast(v16bf, ta0);
        const v16bf af1 = __builtin_bit_cast(v16bf, ta1);
#pragma unroll
        for (int j = 0; j < 4; ++j) {
            U32x8 tb;
            tb.lo = *(const uint4*)(Bp[j] + kb);
            tb.hi = *(const uint4*)(Bp[j] + kb + 16);
            const v16bf bf = __builtin_bit_cast(v16bf, tb);
            acc[0][j] = __builtin_amdgcn_wmma_f32_16x16x32_bf16(
                false, af0, false, bf, (short)0, acc[0][j], false, false);
            acc[1][j] = __builtin_amdgcn_wmma_f32_16x16x32_bf16(
                false, af1, false, bf, (short)0, acc[1][j], false, false);
        }
    }

#pragma unroll
    for (int rs = 0; rs < 2; ++rs) {
#pragma unroll
        for (int j = 0; j < 4; ++j) {
            const int col = tn * 64 + j * 16 + laneLo;
            const float bv = doBias ? bias[col] : 0.f;
#pragma unroll
            for (int r = 0; r < 8; ++r) {
                const int row = tm * 32 + rs * 16 + laneHi * 8 + r;  // C/D: vgpr r -> M=laneHi*8+r
                float v = acc[rs][j][r] + bv;
                if (doRelu) v = fmaxf(v, 0.f);
                if (mode == 0)      outF[(size_t)row * ldc + col] = v;
                else if (mode == 1) outB[(size_t)row * ldc + col] = f2bf(v);
                else                outF[(size_t)(row >> 7) * ((size_t)N * T2)
                                         + (size_t)col * T2 + (row & (T2 - 1))] = v;
            }
        }
    }
}

// ------------------------------------------- fused scores + softmax + head-mean
__global__ void scores_softmax_kernel(const float* __restrict__ U,
                                      const float* __restrict__ Vt,
                                      const float* __restrict__ w2,
                                      const float* __restrict__ b2,
                                      unsigned short* __restrict__ attnB) {
    const int bt = blockIdx.x;            // b*T1 + t
    const int b  = bt >> 7;
    const int t  = bt & (T1 - 1);
    const int s  = threadIdx.x;           // 0..127

    __shared__ float ush[HD];
    __shared__ float wsh[HD];
    __shared__ float red[T2];

    float acc = 0.f;
    for (int n = 0; n < NHEAD; ++n) {
        const float* Urow = U + ((size_t)(n * BB + b) * T1 + t) * HD;
        const float* w2n  = w2 + n * HD;
        for (int h = s; h < HD; h += T2) { ush[h] = Urow[h]; wsh[h] = w2n[h]; }
        __syncthreads();

        const float* Vhb = Vt + (size_t)(n * BB + b) * HD * T2;
        float sc = b2[n];
#pragma unroll 4
        for (int h = 0; h < HD; ++h) {
            float v = ush[h] + Vhb[(size_t)h * T2 + s];
            sc = fmaf(fmaxf(v, 0.f), wsh[h], sc);
        }

        red[s] = sc; __syncthreads();
        for (int off = 64; off > 0; off >>= 1) {
            if (s < off) red[s] = fmaxf(red[s], red[s + off]);
            __syncthreads();
        }
        const float mx = red[0]; __syncthreads();
        const float e = __expf(sc - mx);
        red[s] = e; __syncthreads();
        for (int off = 64; off > 0; off >>= 1) {
            if (s < off) red[s] += red[s + off];
            __syncthreads();
        }
        const float sum = red[0]; __syncthreads();
        acc += e / sum;
    }
    attnB[(size_t)bt * T2 + s] = f2bf(acc * (1.0f / NHEAD));
}

// ------------------------------------------------------- layernorm + mean_t
__global__ void ln_mean_kernel(const float* __restrict__ x,
                               const float* __restrict__ g,
                               const float* __restrict__ be,
                               float* __restrict__ out) {
    const int b = blockIdx.x;             // 4 blocks
    const int d = threadIdx.x;            // 768 threads
    __shared__ float red[DD];
    float acc = 0.f;
    for (int t = 0; t < T1; ++t) {
        const float v = x[((size_t)b * T1 + t) * DD + d];
        red[d] = v; __syncthreads();
        for (int off = 512; off > 0; off >>= 1) {
            if (d < off && d + off < DD) red[d] += red[d + off];
            __syncthreads();
        }
        const float mu = red[0] * (1.0f / DD); __syncthreads();
        const float c = v - mu;
        red[d] = c * c; __syncthreads();
        for (int off = 512; off > 0; off >>= 1) {
            if (d < off && d + off < DD) red[d] += red[d + off];
            __syncthreads();
        }
        const float var = red[0] * (1.0f / DD); __syncthreads();
        acc += c * rsqrtf(var + LN_EPS) * g[d] + be[d];
    }
    out[(size_t)b * DD + d] = acc * (1.0f / T1);
}

// ---------------------------------------------------------------- launch
extern "C" void kernel_launch(void* const* d_in, const int* in_sizes, int n_in,
                              void* d_out, int out_size, void* d_ws, size_t ws_size,
                              hipStream_t stream) {
    const float* speech = (const float*)d_in[0];
    const float* phon   = (const float*)d_in[1];
    const float* W1     = (const float*)d_in[2];
    const float* b1     = (const float*)d_in[3];
    const float* w2     = (const float*)d_in[4];
    const float* b2     = (const float*)d_in[5];
    const float* Wf1    = (const float*)d_in[6];
    const float* bf1    = (const float*)d_in[7];
    const float* Wf2    = (const float*)d_in[8];
    const float* bf2    = (const float*)d_in[9];
    const float* ln_g   = (const float*)d_in[10];
    const float* ln_b   = (const float*)d_in[11];
    float* out = (float*)d_out;

    char* w = (char*)d_ws;
    size_t off = 0;
    auto alloc = [&](size_t bytes) -> void* {
        void* p = w + off;
        off = (off + bytes + 255) & ~(size_t)255;
        return p;
    };

    const int nSp = BB * T1 * DD;                       // 393216

    unsigned short* speechB = (unsigned short*)alloc((size_t)nSp * 2);
    unsigned short* phonB   = (unsigned short*)alloc((size_t)nSp * 2);
    unsigned short* W1t     = (unsigned short*)alloc((size_t)NHEAD * HD * 2 * DD * 2); // [n][h][2D]
    unsigned short* Wf1t    = (unsigned short*)alloc((size_t)2 * HD * 2 * DD * 2);     // [2HD][2D]
    unsigned short* Wf2t    = (unsigned short*)alloc((size_t)DD * 2 * HD * 2);         // [D][2HD]
    unsigned short* phonT   = (unsigned short*)alloc((size_t)BB * DD * T2 * 2);        // [b][d][s]
    float*          U       = (float*)alloc((size_t)NHEAD * BB * T1 * HD * 4);
    float*          Vt      = (float*)alloc((size_t)NHEAD * BB * HD * T2 * 4);
    unsigned short* attnB   = (unsigned short*)alloc((size_t)BB * T1 * T2 * 2);
    unsigned short* fusedB  = (unsigned short*)alloc((size_t)BB * T1 * 2 * DD * 2);
    unsigned short* hB      = (unsigned short*)alloc((size_t)BB * T1 * 2 * HD * 2);
    float*          outtok  = (float*)alloc((size_t)BB * T1 * DD * 4);

    // A-side converts (row-major bf16), vectorized x4
    f2bf4_kernel<<<(nSp / 4 + 255) / 256, 256, 0, stream>>>((const float4*)speech, (ushort4*)speechB, nSp / 4);
    f2bf4_kernel<<<(nSp / 4 + 255) / 256, 256, 0, stream>>>((const float4*)phon,   (ushort4*)phonB,   nSp / 4);

    // B-side transposed converts: Bt[N][K]
    dim3 tb(32, 8);
    transpose_f2bf<<<dim3(DD / 32, 2 * DD / 32, NHEAD), tb, 0, stream>>>(
        W1, W1t, 2 * DD, HD, (long long)2 * DD * HD, (long long)HD * 2 * DD);
    transpose_f2bf<<<dim3(2 * HD / 32, 2 * DD / 32, 1), tb, 0, stream>>>(
        Wf1, Wf1t, 2 * DD, 2 * HD, 0, 0);
    transpose_f2bf<<<dim3(DD / 32, 2 * HD / 32, 1), tb, 0, stream>>>(
        Wf2, Wf2t, 2 * HD, DD, 0, 0);
    transpose_f2bf<<<dim3(DD / 32, T2 / 32, BB), tb, 0, stream>>>(
        phon, phonT, T2, DD, (long long)T2 * DD, (long long)DD * T2);

    // U[n][b*T1+t][h] = speech @ Wx[n] + b1[n]   (mode 0 + bias)
    wmma_gemm_bf16<<<dim3(24, 1, NHEAD), 256, 0, stream>>>(
        speechB, W1t, b1, U, nullptr,
        BB * T1, HD, DD, DD, 2 * DD, HD,
        0, (long long)HD * 2 * DD, HD, (long long)BB * T1 * HD, 0, 1, 0);

    // Vt[n][b][h][s] = phon @ Wy[n]  (mode 2: transposed store; Wy = cols 768.. of W1t rows)
    wmma_gemm_bf16<<<dim3(24, 1, NHEAD), 256, 0, stream>>>(
        phonB, W1t + DD, nullptr, Vt, nullptr,
        BB * T2, HD, DD, DD, 2 * DD, T2,
        0, (long long)HD * 2 * DD, 0, (long long)BB * HD * T2, 2, 0, 0);

    // scores -> softmax -> head mean -> attn (bf16)
    scores_softmax_kernel<<<BB * T1, T2, 0, stream>>>(U, Vt, w2, b2, attnB);

    // fused[:, 0:768] = speech
    pack_speech_kernel<<<(nSp / 4 + 255) / 256, 256, 0, stream>>>((const uint2*)speechB, (uint2*)fusedB);

    // fused[:, 768:1536] = attn @ phoneme  (mode 1 bf16 out; B = phonT per batch)
    wmma_gemm_bf16<<<dim3(6, 1, BB), 256, 0, stream>>>(
        attnB, phonT, nullptr, nullptr, fusedB + DD,
        T1, DD, T2, T2, T2, 2 * DD,
        (long long)T1 * T2, (long long)DD * T2, 0, (long long)T1 * 2 * DD, 1, 0, 0);

    // h = relu(fused @ Wf1 + bf1)
    wmma_gemm_bf16<<<dim3(48, 1, 1), 256, 0, stream>>>(
        fusedB, Wf1t, bf1, nullptr, hB,
        BB * T1, 2 * HD, 2 * DD, 2 * DD, 2 * DD, 2 * HD,
        0, 0, 0, 0, 1, 1, 1);

    // outtok = h @ Wf2 + bf2
    wmma_gemm_bf16<<<dim3(24, 1, 1), 256, 0, stream>>>(
        hB, Wf2t, bf2, outtok, nullptr,
        BB * T1, DD, 2 * HD, 2 * HD, 2 * HD, DD,
        0, 0, 0, 0, 0, 1, 0);

    // layernorm + mean over T1
    ln_mean_kernel<<<BB, DD, 0, stream>>>(outtok, ln_g, ln_b, out);
}
